// BiLSTM_58351425683854
// MI455X (gfx1250) — matrix operand order repaired and checked by reading, not compiled, not run on hardware
//
#include <hip/hip_runtime.h>

typedef _Float16 half_t;
typedef __attribute__((ext_vector_type(16))) _Float16 v16h;
typedef __attribute__((ext_vector_type(8)))  _Float16 v8h;
typedef __attribute__((ext_vector_type(4)))  _Float16 v4h;
typedef __attribute__((ext_vector_type(8)))  float    v8f;
typedef __attribute__((ext_vector_type(4)))  float    v4f;

#define SEQ_  100
#define B_    30
#define H_    512
#define MTOT  (SEQ_ * B_)      // 3000
#define MPAD  3072             // padded to a multiple of BM -> branch-free GEMM

#define BM 128
#define BN 64
#define BK 32
#define NKT (H_ / BK)          // 16 K-iterations
#define LDA 40                 // A-tile LDS row stride (halves): 80B, 16B-aligned, conflict-free frags
#define LDB 520                // B-panel LDS row stride (halves): 1040B ≡ 4 dwords mod 64 banks
#define LES 72                 // epilogue LDS scratch row stride (halves): 144B, 16B-aligned

__device__ __forceinline__ v16h cat8(v8h lo, v8h hi) {
  v16h r;
#pragma unroll
  for (int i = 0; i < 8; ++i) { r[i] = lo[i]; r[i + 8] = hi[i]; }
  return r;
}

__device__ __forceinline__ float sigmoid_f(float x) {
  return 1.0f / (1.0f + exp2f(x * -1.44269504088896340736f));
}

// Y[m,n] = sigmoid( sum_k (P[m,k]+Q[m,k]) * W[n,k] + bias[n] )
// P,Q,Y are MPAD x 512 f16 (rows >= 3000 are don't-care padding), W is 512x512 f16.
// Launched with gridDim.z == 1 (set 0 only) or 2 (blockIdx.z selects operand set):
// the two operand sets are independent GEMMs, merged into one launch to double
// resident workgroups during the independent phases of the interaction DAG.
__global__ __launch_bounds__(256) void gemm_bias_sigmoid(
    const half_t* __restrict__ P0, const half_t* __restrict__ Q0,
    const half_t* __restrict__ W0, const float* __restrict__ bias0,
    half_t* __restrict__ Y0,
    const half_t* __restrict__ P1, const half_t* __restrict__ Q1,
    const half_t* __restrict__ W1, const float* __restrict__ bias1,
    half_t* __restrict__ Y1)
{
  __shared__ half_t Bs[BN * LDB];        // whole weight panel: 64 rows x 512 halves (padded)
  __shared__ half_t As[2][BM * LDA];     // ping-pong (P+Q) tile, row-major [m][k]

  // uniform (per-block) operand-set select
  const bool g1 = (blockIdx.z != 0);
  const half_t* __restrict__ P    = g1 ? P1    : P0;
  const half_t* __restrict__ Q    = g1 ? Q1    : Q0;
  const half_t* __restrict__ W    = g1 ? W1    : W0;
  const float*  __restrict__ bias = g1 ? bias1 : bias0;
  half_t*       __restrict__ Y    = g1 ? Y1    : Y0;

  const int tid  = threadIdx.x;
  const int lane = tid & 31;
  const int wave = tid >> 5;             // 0..7
  const int wm   = wave & 3;             // 4 waves along M
  const int wn   = wave >> 2;            // 2 waves along N
  const int bm   = blockIdx.x * BM;
  const int bn   = blockIdx.y * BN;
  const int sel  = lane >> 4;
  const int l16  = lane & 15;

  // ---- async preload of the entire B panel (64 x 512 f16 = 64KB) into LDS.
  // Pure copy -> CDNA5 async-to-LDS path (no VGPR staging), tracked by ASYNCcnt.
  {
    const half_t* wbase = W + (size_t)bn * H_;
#pragma unroll
    for (int it = 0; it < 16; ++it) {
      const int c   = tid + it * 256;     // 0..4095 chunks of 8 halves
      const int row = c >> 6;             // 0..63
      const int k8  = (c & 63) * 8;       // half offset within row
      unsigned int  ldsb = (unsigned int)(unsigned long long)(&Bs[row * LDB + k8]);
      unsigned long long ga = (unsigned long long)(wbase + (size_t)row * H_ + k8);
      asm volatile("global_load_async_to_lds_b128 %0, %1, off"
                   :: "v"(ldsb), "v"(ga) : "memory");
    }
  }

  // ---- A staging geometry: 128 rows x 4 chunks(8 halves) = 512 chunks, 2 per thread
  int arow[2], ack[2];
  size_t abase[2];
#pragma unroll
  for (int it = 0; it < 2; ++it) {
    const int cid = tid + it * 256;
    arow[it]  = cid >> 2;
    ack[it]   = (cid & 3) * 8;
    abase[it] = (size_t)(bm + arow[it]) * H_ + ack[it];
  }

  // ---- prologue: stage A tile for kt=0 into buffer 0 (no guards: M padded)
  {
    v8h pr[2], qr[2];
#pragma unroll
    for (int it = 0; it < 2; ++it) {
      pr[it] = *(const v8h*)(P + abase[it]);
      qr[it] = *(const v8h*)(Q + abase[it]);
    }
#pragma unroll
    for (int it = 0; it < 2; ++it)
      *(v8h*)(&As[0][arow[it] * LDA + ack[it]]) = pr[it] + qr[it];
  }
  asm volatile("s_wait_asynccnt 0x0" ::: "memory");   // B panel resident (this wave's DMAs)
  __syncthreads();

  v8f acc[2][2] = {};                    // [m-tile][n-tile] f32 accumulators

  for (int kt = 0; kt < NKT; ++kt) {
    const int cur   = kt & 1;
    const int kk    = kt * BK;
    const bool more = (kt + 1) < NKT;

    // -- issue next tile's global loads early (overlaps ds_load + WMMA below)
    v8h pn[2], qn[2];
    if (more) {
#pragma unroll
      for (int it = 0; it < 2; ++it) {
        pn[it] = *(const v8h*)(P + abase[it] + kk + BK);
        qn[it] = *(const v8h*)(Q + abase[it] + kk + BK);
      }
    }

    // -- fragments (contiguous k: lane half sel holds k = sel*16 .. +15 for A and B)
    v16h af[2], bf[2];
#pragma unroll
    for (int mt = 0; mt < 2; ++mt) {
      const int m = wm * 32 + mt * 16 + l16;
      const half_t* ap = &As[cur][m * LDA + sel * 16];
      v8h lo = *(const v8h*)(ap);        // K = sel*16 .. +7
      v8h hi = *(const v8h*)(ap + 8);    // K = sel*16+8 .. +15
      af[mt] = cat8(lo, hi);
    }
#pragma unroll
    for (int nt = 0; nt < 2; ++nt) {
      const int n = wn * 32 + nt * 16 + l16;
      bf[nt] = *(const v16h*)(&Bs[n * LDB + kk + sel * 16]);
    }
#pragma unroll
    for (int mt = 0; mt < 2; ++mt)
#pragma unroll
      for (int nt = 0; nt < 2; ++nt)
        acc[mt][nt] = __builtin_amdgcn_wmma_f32_16x16x32_f16(
            false, af[mt], false, bf[nt],
            (short)0, acc[mt][nt], false, false);

    // -- commit next tile into the other buffer
    if (more) {
#pragma unroll
      for (int it = 0; it < 2; ++it)
        *(v8h*)(&As[cur ^ 1][arow[it] * LDA + ack[it]]) = pn[it] + qn[it];
    }
    __syncthreads();
  }
  // (loop ended with a barrier: all LDS traffic drained; safe to reuse As as scratch)

  // ---- epilogue: bias + sigmoid into LDS scratch, then coalesced b128 stores
  half_t* Ys = &As[0][0];                // 128 x LES halves = 9216 <= 10240 avail
#pragma unroll
  for (int nt = 0; nt < 2; ++nt) {
    const int nl = wn * 32 + nt * 16 + l16;          // local n: 0..63
    const float bb = bias[bn + nl];
#pragma unroll
    for (int mt = 0; mt < 2; ++mt) {
#pragma unroll
      for (int r = 0; r < 8; ++r) {
        const int ml = wm * 32 + mt * 16 + sel * 8 + r;  // local m: 0..127
        Ys[ml * LES + nl] = (half_t)sigmoid_f(acc[mt][nt][r] + bb);
      }
    }
  }
  __syncthreads();
#pragma unroll
  for (int it = 0; it < 4; ++it) {
    const int c   = tid + it * 256;      // 0..1023 chunks of 8 halves
    const int row = c >> 3;              // 0..127
    const int ck  = (c & 7) * 8;         // 0..56
    v8h v = *(const v8h*)(&Ys[row * LES + ck]);
    *(v8h*)(Y + (size_t)(bm + row) * H_ + bn + ck) = v;
  }
}

__global__ __launch_bounds__(256) void f32_to_f16_k(const float* __restrict__ in,
                                                    half_t* __restrict__ out, int n4) {
  int i = blockIdx.x * 256 + threadIdx.x;
  if (i < n4) {
    v4f v = *(const v4f*)(in + 4 * (size_t)i);
    v4h h;
#pragma unroll
    for (int j = 0; j < 4; ++j) h[j] = (half_t)v[j];
    *(v4h*)(out + 4 * (size_t)i) = h;
  }
}

__global__ __launch_bounds__(256) void zero_f16_k(half_t* __restrict__ p, int n8) {
  int i = blockIdx.x * 256 + threadIdx.x;
  if (i < n8) {
    v8h z = {};
    *(v8h*)(p + 8 * (size_t)i) = z;
  }
}

__global__ __launch_bounds__(256) void final_avg_k(const half_t* __restrict__ hf,
                                                   const half_t* __restrict__ hb,
                                                   float* __restrict__ out, int n4) {
  int i = blockIdx.x * 256 + threadIdx.x;
  if (i < n4) {
    v4h a = *(const v4h*)(hf + 4 * (size_t)i);
    v4h b = *(const v4h*)(hb + 4 * (size_t)i);
    v4f o;
#pragma unroll
    for (int j = 0; j < 4; ++j) o[j] = ((float)a[j] + (float)b[j]) * 0.5f;
    *(v4f*)(out + 4 * (size_t)i) = o;
  }
}

extern "C" void kernel_launch(void* const* d_in, const int* in_sizes, int n_in,
                              void* d_out, int out_size, void* d_ws, size_t ws_size,
                              hipStream_t stream) {
  (void)in_sizes; (void)n_in; (void)out_size; (void)ws_size;

  // setup_inputs() dict order: inputs, W1..W4, b1..b4, (LSTM params: dead code)
  const float* inputs = (const float*)d_in[0];
  const float* Wf[4]  = {(const float*)d_in[1], (const float*)d_in[2],
                         (const float*)d_in[3], (const float*)d_in[4]};
  const float* bf[4]  = {(const float*)d_in[5], (const float*)d_in[6],
                         (const float*)d_in[7], (const float*)d_in[8]};

  char* ws = (char*)d_ws;
  size_t off = 0;
  auto take = [&](size_t bytes) -> void* {
    void* p = ws + off;
    off += (bytes + 255) & ~(size_t)255;
    return p;
  };

  const size_t actB = (size_t)MPAD * H_ * sizeof(half_t);  // 3 MB each (padded)
  half_t* x0  = (half_t*)take(actB);
  half_t* x1  = (half_t*)take(actB);
  half_t* x2  = (half_t*)take(actB);
  half_t* hf  = (half_t*)take(actB);
  half_t* hb  = (half_t*)take(actB);
  half_t* hf2 = (half_t*)take(actB);
  half_t* hb2 = (half_t*)take(actB);
  half_t* Wh[4];
  for (int i = 0; i < 4; ++i) Wh[i] = (half_t*)take((size_t)H_ * H_ * sizeof(half_t));

  const int nAct  = MTOT * H_;           // 1,536,000 real elements
  const int nActP = MPAD * H_;           // 1,572,864 padded elements
  const int nPad  = nActP - nAct;        // 36,864 pad elements
  const int nW    = H_ * H_;             // 262,144

  // prep: f32->f16 conversion of inputs + zero pad rows; weights; zero h states (padded)
  f32_to_f16_k<<<(nAct / 4 + 255) / 256, 256, 0, stream>>>(inputs, x0, nAct / 4);
  zero_f16_k<<<(nPad / 8 + 255) / 256, 256, 0, stream>>>(x0 + nAct, nPad / 8);
  for (int i = 0; i < 4; ++i)
    f32_to_f16_k<<<(nW / 4 + 255) / 256, 256, 0, stream>>>(Wf[i], Wh[i], nW / 4);
  zero_f16_k<<<(nActP / 8 + 255) / 256, 256, 0, stream>>>(hf, nActP / 8);
  zero_f16_k<<<(nActP / 8 + 255) / 256, 256, 0, stream>>>(hb, nActP / 8);

  dim3 grid1(MPAD / BM, H_ / BN, 1);     // 24 x 8 x 1
  dim3 grid2(MPAD / BM, H_ / BN, 2);     // 24 x 8 x 2 (two independent GEMMs)
  dim3 block(256);

#define GEMM1(P, Q, W, B, Y) \
  gemm_bias_sigmoid<<<grid1, block, 0, stream>>>(P, Q, W, B, Y, P, Q, W, B, Y)
#define GEMM2(P0, Q0, W0, B0, Y0, P1, Q1, W1, B1, Y1) \
  gemm_bias_sigmoid<<<grid2, block, 0, stream>>>(P0, Q0, W0, B0, Y0, P1, Q1, W1, B1, Y1)

  for (int t = 0; t < SEQ_; ++t) {
    // interaction, iteration 1
    GEMM1(x0,  hf,  Wh[0], bf[0], x1);                                   // x1
    GEMM2(hb,  x1,  Wh[1], bf[1], hb2,  x1, hf,  Wh[2], bf[2], hf2);     // hb2 || hf2
    GEMM1(hb2, x1,  Wh[3], bf[3], x2);                                   // x2
    // interaction, iteration 2 (its final x2 is dead -> skipped)
    GEMM1(x2,  hf2, Wh[0], bf[0], x1);                                   // x1'
    GEMM2(hb2, x1,  Wh[1], bf[1], hb,   x1, hf2, Wh[2], bf[2], hf);      // hb' || hf'
    // LSTM cells: dead code w.r.t. output (cf/cb never reach hf/hb) -> skipped
  }
#undef GEMM1
#undef GEMM2

  final_avg_k<<<(nAct / 4 + 255) / 256, 256, 0, stream>>>(hf, hb, (float*)d_out, nAct / 4);
}
